// Rollout_7524782702956
// MI455X (gfx1250) — matrix operand order repaired and checked by reading, not compiled, not run on hardware
//
#include <hip/hip_runtime.h>

typedef _Float16 v16h __attribute__((ext_vector_type(16)));
typedef _Float16 v8h  __attribute__((ext_vector_type(8)));
typedef float    v8f  __attribute__((ext_vector_type(8)));

// Explicit global-address-space pointee types so laundered weight loads lower
// to GLOBAL_LOAD_B128, not FLAT_LOAD_B128.
typedef __attribute__((address_space(1))) const char GByte;
typedef __attribute__((address_space(1))) const v8h  GV8h;

#define BATCH 1024
#define SEQ   128
#define HID   256
#define GIVEN 32
#define GATES (4 * HID)   // 1024

#define LOG2E 1.4426950408889634f

// Native-transcendental activations: v_exp_f32 + v_rcp_f32 (TRANS32 pipe),
// ~4-5 VALU each instead of ~18 for libm tanhf/expf expansions.
__device__ __forceinline__ float fast_sigmoid(float x) {
    float e = __builtin_amdgcn_exp2f(-x * LOG2E);
    return __builtin_amdgcn_rcpf(1.0f + e);
}
__device__ __forceinline__ float fast_tanh(float x) {
    float e = __builtin_amdgcn_exp2f(-2.0f * LOG2E * x);
    return 2.0f * __builtin_amdgcn_rcpf(1.0f + e) - 1.0f;
}

// ---------------------------------------------------------------------------
// Prep kernel: W_hh (f32, [4H, H] row-major = [n][k]) -> f16 workspace with the
// SAME [n][k] layout (so B fragments are two contiguous 8-f16 runs per lane),
// plus fused bias[n] = b_ih[n] + b_hh[n].
// ---------------------------------------------------------------------------
__global__ void prep_weights(const float* __restrict__ Whh,
                             const float* __restrict__ bih,
                             const float* __restrict__ bhh,
                             _Float16* __restrict__ w16,
                             float* __restrict__ bias) {
    int i = blockIdx.x * 256 + threadIdx.x;
    if (i < GATES * HID) w16[i] = (_Float16)Whh[i];
    if (i < GATES)       bias[i] = bih[i] + bhh[i];
}

// ---------------------------------------------------------------------------
// Main kernel: one workgroup = 16 batch rows, 8 waves (256 threads, wave32).
// Wave w owns hidden units [32w, 32w+32): 8 gate tiles of 16 cols
// (4 gates x 2 sub-tiles). h lives in LDS (f16, row-major [m][k]); c lives in
// VGPRs in the WMMA C/D fragment layout. Weights stream from L2 every step;
// the base pointer is laundered through an SGPR asm constraint per step so
// LICM cannot hoist 512 VGPRs worth of weight fragments out of the time loop,
// and accessed via an addrspace(1) pointer so loads select GLOBAL (not FLAT).
// ---------------------------------------------------------------------------
__global__ __launch_bounds__(256)
void lstm_rollout(const float* __restrict__ prefix,   // [B, SEQ]
                  const float* __restrict__ Wih,      // [4H]
                  const _Float16* __restrict__ w16,   // [4H, H] f16
                  const float* __restrict__ bias,     // [4H] fused
                  const float* __restrict__ Wout,     // [H]
                  const float* __restrict__ bout_p,   // [1]
                  float* __restrict__ out)            // [B, SEQ]
{
    __shared__ __align__(16) _Float16 hl[16 * HID];   // h tile, row-major [m][k]
    __shared__ float xl[16];                          // current input per row
    __shared__ float red[256];                        // dot-product partials

    const int tid  = threadIdx.x;
    const int wv   = tid >> 5;        // wave id 0..7
    const int lane = tid & 31;
    const int lr   = lane & 15;       // fragment row/col index
    const int kh   = lane >> 4;       // k-half select (0/1)
    const int b0   = blockIdx.x * 16; // batch tile base

    // h0 = 0
    for (int i = tid; i < 16 * HID; i += 256) hl[i] = (_Float16)0.0f;

    // out[:, :GIVEN] = prefix[:, :GIVEN]
    for (int i = tid; i < 16 * GIVEN; i += 256) {
        int r = i >> 5;        // GIVEN == 32
        int t = i & 31;
        out[(size_t)(b0 + r) * SEQ + t] = prefix[(size_t)(b0 + r) * SEQ + t];
    }

    // Per-lane constants for this wave's 8 gate tiles: t = gate*2 + sub
    const int hbase = wv * 32;
    float wih_r[8], bias_r[8];
    unsigned int woff[8];   // byte offset of this lane's B-fragment base per tile
#pragma unroll
    for (int t = 0; t < 8; ++t) {
        int g  = t >> 1, s = t & 1;
        int n  = g * HID + hbase + s * 16 + lr;       // gate column for this lane
        wih_r[t]  = Wih[n];
        bias_r[t] = bias[n];
        // lane holds K = k0 + 8*kh .. +7 and k0 + 16 + 8*kh .. +7 of column n
        woff[t] = (unsigned int)((n * HID + kh * 8) * 2);
    }

    // W_out slice for the cooperative output dot (thread tid -> k = (tid&15)*16 + j)
    float woutr[16];
    {
        int kl = tid & 15;
#pragma unroll
        for (int j = 0; j < 16; ++j) woutr[j] = Wout[kl * 16 + j];
    }
    const float bout = bout_p[0];

    const v8f vzero = {0.f, 0.f, 0.f, 0.f, 0.f, 0.f, 0.f, 0.f};
    v8f cst[2];                                       // c state (2 sub-tiles)
    cst[0] = vzero;
    cst[1] = vzero;

    // A fragment base: lane holds row m = lr, runs at K = k0 + 8*kh and +16
    const _Float16* ab = &hl[lr * HID + kh * 8];

    // Laundered weight base (kept in an SGPR pair, made opaque each step).
    unsigned long long wbase = (unsigned long long)w16;

    for (int step = 0; step < SEQ; ++step) {
        const bool warm = (step < GIVEN);
        if (warm && tid < 16)
            xl[tid] = prefix[(size_t)(b0 + tid) * SEQ + step];
        __syncthreads();   // publish xl (warm) / previous h + xl (gen)

        // Defeat LICM: weight loads below must re-issue every step (from L2),
        // not be hoisted into ~512 live VGPRs + scratch spills.
        asm volatile("" : "+s"(wbase));
        GByte* wb = (GByte*)wbase;   // addrspace(1) -> GLOBAL_LOAD

        // per-lane x values for the 8 fragment rows
        float xv[8];
#pragma unroll
        for (int e = 0; e < 8; ++e) xv[e] = xl[e + 8 * kh];

        // gates = h @ W_hh.T  (K = 256, 8 WMMAs per tile, 8 tiles per wave)
        v8f acc[8];
#pragma unroll
        for (int t = 0; t < 8; ++t) acc[t] = vzero;

#pragma unroll
        for (int k0 = 0; k0 < HID; k0 += 32) {
            v8h alo = *(const v8h*)(ab + k0);
            v8h ahi = *(const v8h*)(ab + k0 + 16);
            v16h af = __builtin_shufflevector(alo, ahi,
                0, 1, 2, 3, 4, 5, 6, 7, 8, 9, 10, 11, 12, 13, 14, 15);
#pragma unroll
            for (int t = 0; t < 8; ++t) {
                unsigned int o = woff[t] + (unsigned int)(k0 * 2);
                v8h blo = *(GV8h*)(wb + o);
                v8h bhi = *(GV8h*)(wb + o + 32);
                v16h bf = __builtin_shufflevector(blo, bhi,
                    0, 1, 2, 3, 4, 5, 6, 7, 8, 9, 10, 11, 12, 13, 14, 15);
                acc[t] = __builtin_amdgcn_wmma_f32_16x16x32_f16(
                    false, af, false, bf, (short)0, acc[t], false, false);
            }
        }

        // gates += x * W_ih[n] + (b_ih[n] + b_hh[n])
#pragma unroll
        for (int t = 0; t < 8; ++t)
#pragma unroll
            for (int e = 0; e < 8; ++e)
                acc[t][e] += xv[e] * wih_r[t] + bias_r[t];

        __syncthreads();   // everyone done reading hl; safe to overwrite

        // LSTM cell update in fragment layout; write new h (f16) to LDS
#pragma unroll
        for (int s = 0; s < 2; ++s) {
            int k = hbase + s * 16 + lr;
#pragma unroll
            for (int e = 0; e < 8; ++e) {
                float ig = fast_sigmoid(acc[0 + s][e]);
                float fg = fast_sigmoid(acc[2 + s][e]);
                float gg = fast_tanh(acc[4 + s][e]);
                float og = fast_sigmoid(acc[6 + s][e]);
                float c  = fg * cst[s][e] + ig * gg;
                cst[s][e] = c;
                int m = e + 8 * kh;
                hl[m * HID + k] = (_Float16)(og * fast_tanh(c));
            }
        }
        __syncthreads();   // publish new h

        if (!warm) {
            // nxt = h @ W_out.T + b_out ; feed back as next x, store output.
            int row = tid >> 4, kl = tid & 15;
            const _Float16* hp = &hl[row * HID + kl * 16];
            float p = 0.0f;
#pragma unroll
            for (int j = 0; j < 16; ++j) p += (float)hp[j] * woutr[j];
            red[tid] = p;      // red[row*16 + kl]
            __syncthreads();
            if (tid < 16) {
                float s = bout;
#pragma unroll
                for (int j = 0; j < 16; ++j) s += red[tid * 16 + j];
                out[(size_t)(b0 + tid) * SEQ + step] = s;
                xl[tid] = s;   // cur for next step
            }
            __syncthreads();
        }
    }
}

// ---------------------------------------------------------------------------
// Launch
// ---------------------------------------------------------------------------
extern "C" void kernel_launch(void* const* d_in, const int* in_sizes, int n_in,
                              void* d_out, int out_size, void* d_ws, size_t ws_size,
                              hipStream_t stream) {
    (void)in_sizes; (void)n_in; (void)out_size; (void)ws_size;

    const float* prefix = (const float*)d_in[0];   // [B, SEQ, 1]
    const float* Wih    = (const float*)d_in[1];   // [4H, 1]
    const float* Whh    = (const float*)d_in[2];   // [4H, H]
    const float* bih    = (const float*)d_in[3];   // [4H]
    const float* bhh    = (const float*)d_in[4];   // [4H]
    const float* Wout   = (const float*)d_in[5];   // [1, H]
    const float* bout   = (const float*)d_in[6];   // [1]
    float*       out    = (float*)d_out;           // [B, SEQ, 1]

    _Float16* w16  = (_Float16*)d_ws;                                  // 512 KB
    float*    bias = (float*)((char*)d_ws + (size_t)GATES * HID * 2);  // 4 KB

    prep_weights<<<(GATES * HID + 255) / 256, 256, 0, stream>>>(
        Whh, bih, bhh, w16, bias);

    lstm_rollout<<<BATCH / 16, 256, 0, stream>>>(
        prefix, Wih, w16, bias, Wout, bout, out);
}
